// MLPSinkhorn_4398046511618
// MI455X (gfx1250) — matrix operand order repaired and checked by reading, not compiled, not run on hardware
//
#include <hip/hip_runtime.h>
#include <hip/hip_bf16.h>
#include <math.h>

typedef __attribute__((ext_vector_type(16))) __bf16 v16bf;
typedef __attribute__((ext_vector_type(8)))  float  v8f;
typedef unsigned short ushort_t;
typedef unsigned int   uint_t;

__device__ __forceinline__ ushort_t f2bf(float f) {
    uint_t x = __float_as_uint(f);
    uint_t r = (x + 0x7FFFu + ((x >> 16) & 1u)) >> 16;  // round-to-nearest-even
    return (ushort_t)r;
}

union ABFrag { v16bf v; uint4 q[2]; };

// ---------------- conversion / transpose helpers ----------------

__global__ void k_f32_to_bf16(const float* __restrict__ src, ushort_t* __restrict__ dst, int n) {
    int i = blockIdx.x * blockDim.x + threadIdx.x;
    if (i < n) dst[i] = f2bf(src[i]);
}

// W[K,N] fp32 (row-major) -> Wt[N,K] bf16
__global__ void k_wtrans(const float* __restrict__ W, ushort_t* __restrict__ Wt, int K, int N) {
    int i = blockIdx.x * blockDim.x + threadIdx.x;
    if (i < K * N) {
        int k = i / N, n = i - k * N;
        Wt[(size_t)n * K + k] = f2bf(W[i]);
    }
}

__global__ void k_fill(float* __restrict__ p, float val, int n) {
    int i = blockIdx.x * blockDim.x + threadIdx.x;
    if (i < n) p[i] = val;
}

// row-wise squared L2 norm of fp32 [R, D] (one wave per row)
__global__ void k_rownorm(const float* __restrict__ X, float* __restrict__ out, int R, int D) {
    int lane = threadIdx.x & 31;
    int row  = blockIdx.x * (blockDim.x >> 5) + (threadIdx.x >> 5);
    if (row >= R) return;
    float s = 0.f;
    for (int c = lane; c < D; c += 32) { float t = X[(size_t)row * D + c]; s += t * t; }
    for (int off = 16; off; off >>= 1) s += __shfl_xor(s, off, 32);
    if (lane == 0) out[row] = s;
}

// ---------------- WMMA bf16 GEMM with fused epilogue ----------------
// C[M,N] = A[M,K] * Bt[N,K]^T   (A row-major bf16, Bt = B transposed, bf16)
// EPI==0 : out = relu((acc + bias - mean) * g*rsqrt(var+eps) + beta)  (BN+ReLU)
// EPI==1 : out = acc + bias                                          (final fc)
// Wave computes one 16x16 tile; 8 waves/block -> 32(M) x 64(N) block tile.
template <int EPI>
__global__ void __launch_bounds__(256) k_gemm_bf16(
    const ushort_t* __restrict__ A, const ushort_t* __restrict__ Bt,
    int M, int N, int K,
    const float* __restrict__ bias, const float* __restrict__ g,
    const float* __restrict__ be, const float* __restrict__ mean,
    const float* __restrict__ var,
    ushort_t* __restrict__ Cb, float* __restrict__ Cf)
{
    const int lane  = threadIdx.x & 31;
    const int wv    = threadIdx.x >> 5;
    const int tileM = blockIdx.y * 32 + (wv & 1) * 16;
    const int tileN = blockIdx.x * 64 + (wv >> 1) * 16;
    const int r15   = lane & 15;
    const int row   = tileM + r15;          // A fragment row (M = lane&15 for both halves)
    const int col   = tileN + r15;          // B fragment column
    const int kbA   = (lane < 16) ? 0 : 8;  // 16-bit A: K 0..7/16..23 vs 8..15/24..31
    const int kbB   = (lane < 16) ? 0 : 8;  // 16-bit B: K 0..7/16..23 vs 8..15/24..31
    const ushort_t* Ap = A + (size_t)row * K + kbA;
    const ushort_t* Bp = Bt + (size_t)col * K + kbB;

    v8f acc = {};
    for (int k0 = 0; k0 < K; k0 += 32) {
        ABFrag a, b;
        a.q[0] = *(const uint4*)(Ap + k0);        // K = kbA .. kbA+7
        a.q[1] = *(const uint4*)(Ap + k0 + 16);   // K = kbA+16 .. kbA+23
        b.q[0] = *(const uint4*)(Bp + k0);        // K = kbB .. kbB+7
        b.q[1] = *(const uint4*)(Bp + k0 + 16);   // K = kbB+16 .. kbB+23
        acc = __builtin_amdgcn_wmma_f32_16x16x32_bf16(
            false, a.v, false, b.v, (short)0, acc, false, false);
    }

    float bcol = bias[col];
    float sc = 1.f, sh = 0.f;
    if (EPI == 0) {
        sc = g[col] * rsqrtf(var[col] + 1e-5f);
        sh = be[col] - mean[col] * sc;   // (h-m)*sc + be = h*sc + (be - m*sc)
    }
    const int mOff = (lane >> 4) << 3;   // C/D: VGPR r -> M = r (+8 for upper lanes)
    #pragma unroll
    for (int r = 0; r < 8; ++r) {
        int gr = tileM + mOff + r;
        float h = acc[r] + bcol;
        float val = (EPI == 0) ? fmaxf(h * sc + sh, 0.f) : h;
        Cb[(size_t)gr * N + col] = f2bf(val);
        if (Cf) Cf[(size_t)gr * N + col] = val;
    }
}

// ---------------- cost matrix + Gibbs kernel: K = exp(-cdist(h,y)/eps) ----------------
__global__ void __launch_bounds__(256) k_cost_exp(
    const ushort_t* __restrict__ A, const ushort_t* __restrict__ Bt,
    const float* __restrict__ hh, const float* __restrict__ yy,
    float* __restrict__ Kmat, int M, int N, int K)
{
    const int lane  = threadIdx.x & 31;
    const int wv    = threadIdx.x >> 5;
    const int tileM = blockIdx.y * 32 + (wv & 1) * 16;
    const int tileN = blockIdx.x * 64 + (wv >> 1) * 16;
    const int r15   = lane & 15;
    const int row   = tileM + r15;
    const int col   = tileN + r15;
    const int kbA   = (lane < 16) ? 0 : 8;
    const int kbB   = (lane < 16) ? 0 : 8;
    const ushort_t* Ap = A + (size_t)row * K + kbA;
    const ushort_t* Bp = Bt + (size_t)col * K + kbB;

    v8f acc = {};
    for (int k0 = 0; k0 < K; k0 += 32) {
        ABFrag a, b;
        a.q[0] = *(const uint4*)(Ap + k0);
        a.q[1] = *(const uint4*)(Ap + k0 + 16);
        b.q[0] = *(const uint4*)(Bp + k0);
        b.q[1] = *(const uint4*)(Bp + k0 + 16);
        acc = __builtin_amdgcn_wmma_f32_16x16x32_bf16(
            false, a.v, false, b.v, (short)0, acc, false, false);
    }

    float yyc = yy[col];
    const int mOff = (lane >> 4) << 3;
    #pragma unroll
    for (int r = 0; r < 8; ++r) {
        int gr = tileM + mOff + r;
        float sq = hh[gr] + yyc - 2.f * acc[r];
        float c  = sqrtf(fmaxf(sq, 1e-12f));
        Kmat[(size_t)gr * N + col] = __expf(c * -10.0f);   // exp(-C/0.1)
    }
}

// ---------------- Sinkhorn iterations (L2-resident matvecs) ----------------

// u[i] = aval / dot(K[i,:], v)  -- one wave per row, float4 loads
__global__ void __launch_bounds__(256) k_sink_row(
    const float* __restrict__ Kmat, const float* __restrict__ v,
    float* __restrict__ u, float aval, int Ncols)
{
    int lane = threadIdx.x & 31;
    int row  = blockIdx.x * 8 + (threadIdx.x >> 5);
    const float* kp = Kmat + (size_t)row * Ncols;
    float s = 0.f;
    for (int c = lane * 4; c < Ncols; c += 128) {
        float4 kk = *(const float4*)(kp + c);
        float4 vv = *(const float4*)(v + c);
        s += kk.x * vv.x + kk.y * vv.y + kk.z * vv.z + kk.w * vv.w;
    }
    for (int off = 16; off; off >>= 1) s += __shfl_xor(s, off, 32);
    if (lane == 0) u[row] = aval / s;
}

// v[j] = bval / dot(K[:,j], u)  -- thread j walks a column; coalesced across threads,
// u[i] is a uniform scalar broadcast (s_load). Prefetch keeps rows streaming.
__global__ void __launch_bounds__(256) k_sink_col(
    const float* __restrict__ Kmat, const float* __restrict__ u,
    float* __restrict__ v, float bval, int Nrows, int Ncols)
{
    int j = blockIdx.x * blockDim.x + threadIdx.x;
    const float* kp = Kmat + j;
    float s = 0.f;
    for (int i = 0; i < Nrows; i += 8) {
        if (i + 64 < Nrows) __builtin_prefetch(kp + (size_t)(i + 64) * Ncols, 0, 1);
        #pragma unroll
        for (int t = 0; t < 8; ++t)
            s += kp[(size_t)(i + t) * Ncols] * u[i + t];
    }
    v[j] = bval / s;
}

// pi = u[:,None] * K * v[None,:]  (vectorized, 4 elems/thread)
__global__ void __launch_bounds__(256) k_pi(
    const float* __restrict__ Kmat, const float* __restrict__ u,
    const float* __restrict__ v, float* __restrict__ pi, size_t total4)
{
    size_t t = (size_t)blockIdx.x * blockDim.x + threadIdx.x;
    if (t >= total4) return;
    size_t idx = t * 4;
    size_t i = idx >> 12;            // row (Ncols = 4096)
    int    j = (int)(idx & 4095);
    float  ui = u[i];
    float4 kk = *(const float4*)(Kmat + idx);
    float4 vv = *(const float4*)(v + j);
    float4 o;
    o.x = ui * kk.x * vv.x;
    o.y = ui * kk.y * vv.y;
    o.z = ui * kk.z * vv.z;
    o.w = ui * kk.w * vv.w;
    *(float4*)(pi + idx) = o;
}

// ---------------- host-side orchestration ----------------

extern "C" void kernel_launch(void* const* d_in, const int* in_sizes, int n_in,
                              void* d_out, int out_size, void* d_ws, size_t ws_size,
                              hipStream_t stream) {
    (void)in_sizes; (void)n_in; (void)out_size; (void)ws_size;

    const float* x   = (const float*)d_in[0];
    const float* y   = (const float*)d_in[1];
    const float* W1  = (const float*)d_in[2];  const float* b1  = (const float*)d_in[3];
    const float* g1  = (const float*)d_in[4];  const float* be1 = (const float*)d_in[5];
    const float* m1  = (const float*)d_in[6];  const float* v1  = (const float*)d_in[7];
    const float* W2  = (const float*)d_in[8];  const float* b2  = (const float*)d_in[9];
    const float* g2  = (const float*)d_in[10]; const float* be2 = (const float*)d_in[11];
    const float* m2  = (const float*)d_in[12]; const float* v2  = (const float*)d_in[13];
    const float* W3  = (const float*)d_in[14]; const float* b3  = (const float*)d_in[15];
    const float* g3  = (const float*)d_in[16]; const float* be3 = (const float*)d_in[17];
    const float* m3  = (const float*)d_in[18]; const float* v3  = (const float*)d_in[19];
    const float* W4  = (const float*)d_in[20]; const float* b4  = (const float*)d_in[21];

    const int N = 4096, M = 4096, DIN = 2048;
    const int D1 = 256, D2 = 128, D3 = 64, D4 = 256;

    char* wsp = (char*)d_ws;
    auto carve = [&](size_t bytes) -> char* {
        char* p = wsp; wsp += (bytes + 255) & ~(size_t)255; return p;
    };
    ushort_t* xb  = (ushort_t*)carve((size_t)N * DIN * 2);
    ushort_t* yb  = (ushort_t*)carve((size_t)M * D4  * 2);
    ushort_t* wt1 = (ushort_t*)carve((size_t)DIN * D1 * 2);
    ushort_t* wt2 = (ushort_t*)carve((size_t)D1 * D2 * 2);
    ushort_t* wt3 = (ushort_t*)carve((size_t)D2 * D3 * 2);
    ushort_t* wt4 = (ushort_t*)carve((size_t)D3 * D4 * 2);
    ushort_t* h1  = (ushort_t*)carve((size_t)N * D1 * 2);
    ushort_t* h2  = (ushort_t*)carve((size_t)N * D2 * 2);
    ushort_t* h3  = (ushort_t*)carve((size_t)N * D3 * 2);
    ushort_t* h4b = (ushort_t*)carve((size_t)N * D4 * 2);
    float*    h4f = (float*)carve((size_t)N * D4 * 4);
    float*    hh  = (float*)carve((size_t)N * 4);
    float*    yy  = (float*)carve((size_t)M * 4);
    float*    Km  = (float*)carve((size_t)N * M * 4);   // 64 MB, L2-resident
    float*    uu  = (float*)carve((size_t)N * 4);
    float*    vv  = (float*)carve((size_t)M * 4);

    // fp32 -> bf16 staging (A operands) and transposed bf16 weights (B operands)
    { int n = N * DIN; k_f32_to_bf16<<<(n + 255) / 256, 256, 0, stream>>>(x, xb, n); }
    { int n = M * D4;  k_f32_to_bf16<<<(n + 255) / 256, 256, 0, stream>>>(y, yb, n); }
    { int n = DIN * D1; k_wtrans<<<(n + 255) / 256, 256, 0, stream>>>(W1, wt1, DIN, D1); }
    { int n = D1 * D2;  k_wtrans<<<(n + 255) / 256, 256, 0, stream>>>(W2, wt2, D1, D2); }
    { int n = D2 * D3;  k_wtrans<<<(n + 255) / 256, 256, 0, stream>>>(W3, wt3, D2, D3); }
    { int n = D3 * D4;  k_wtrans<<<(n + 255) / 256, 256, 0, stream>>>(W4, wt4, D3, D4); }

    dim3 blk(256);
    // MLP stack (WMMA bf16 GEMMs, fused BN+ReLU / bias epilogues)
    k_gemm_bf16<0><<<dim3(D1 / 64, N / 32), blk, 0, stream>>>(xb, wt1, N, D1, DIN, b1, g1, be1, m1, v1, h1, nullptr);
    k_gemm_bf16<0><<<dim3(D2 / 64, N / 32), blk, 0, stream>>>(h1, wt2, N, D2, D1,  b2, g2, be2, m2, v2, h2, nullptr);
    k_gemm_bf16<0><<<dim3(D3 / 64, N / 32), blk, 0, stream>>>(h2, wt3, N, D3, D2,  b3, g3, be3, m3, v3, h3, nullptr);
    k_gemm_bf16<1><<<dim3(D4 / 64, N / 32), blk, 0, stream>>>(h3, wt4, N, D4, D3,  b4, nullptr, nullptr, nullptr, nullptr, h4b, h4f);

    // squared row norms for cdist
    k_rownorm<<<N / 8, 256, 0, stream>>>(h4f, hh, N, D4);
    k_rownorm<<<M / 8, 256, 0, stream>>>(y,   yy, M, D4);

    // K = exp(-cdist(h,y)/eps) via WMMA GEMM (h @ y^T) + fused sqrt/exp epilogue
    k_cost_exp<<<dim3(M / 64, N / 32), blk, 0, stream>>>(h4b, yb, hh, yy, Km, N, M, D4);

    // Sinkhorn: 100 iterations of u = a/(K v), v = b/(K^T u)
    const float aval = 1.0f / N, bval = 1.0f / M;
    k_fill<<<(M + 255) / 256, 256, 0, stream>>>(vv, bval, M);
    for (int it = 0; it < 100; ++it) {
        k_sink_row<<<N / 8,  256, 0, stream>>>(Km, vv, uu, aval, M);
        k_sink_col<<<M / 256, 256, 0, stream>>>(Km, uu, vv, bval, N, M);
    }

    // transport plan
    size_t total4 = ((size_t)N * M) / 4;
    k_pi<<<(unsigned)((total4 + 255) / 256), 256, 0, stream>>>(Km, uu, vv, (float*)d_out, total4);
}
